// AttentionSelf_64510408786292
// MI455X (gfx1250) — matrix operand order repaired and checked by reading, compiled, not run on hardware
//
#include <hip/hip_runtime.h>

// ---------------------------------------------------------------------------
// Self-attention (B=4, S=4096, Din=256, Dout=64) for gfx1250 / MI455X.
//   Kernel 1: QKV projection -> bf16 Q,K (row-major) and V^T (per-batch) in ws
//   Kernel 2: flash attention, double-buffered KV tiles staged with
//             GLOBAL_LOAD_ASYNC_TO_LDS_B128 (ASYNCcnt), bf16 WMMA, fp32 accum
// ---------------------------------------------------------------------------

typedef __attribute__((ext_vector_type(16))) __bf16 bf16x16;
typedef __attribute__((ext_vector_type(8)))  float  f32x8;
typedef __attribute__((ext_vector_type(4)))  int    v4i;

typedef __attribute__((address_space(1))) v4i* gv4i_p;   // global int4*
typedef __attribute__((address_space(3))) v4i* lv4i_p;   // LDS int4*

#define DEV static __device__ __forceinline__

static constexpr int   S_LEN    = 4096;
static constexpr int   D_IN     = 256;
static constexpr int   D_OUT    = 64;
static constexpr int   BATCH    = 4;
static constexpr float SCALE    = 0.125f;              // 1/sqrt(64)
static constexpr int   ROWS     = BATCH * S_LEN;       // 16384
static constexpr int   QKV_ELTS = ROWS * D_OUT;        // per matrix, bf16

// ---------------------------------------------------------------------------
// Async global->LDS staging (CDNA5 ASYNCcnt path) with sync fallback.
// ---------------------------------------------------------------------------
#if __has_builtin(__builtin_amdgcn_global_load_async_to_lds_b128)
#define HAS_ASYNC_LDS 1
#else
#define HAS_ASYNC_LDS 0
#endif

DEV void copy16_to_lds(const unsigned short* g, unsigned short* l) {
#if HAS_ASYNC_LDS
  __builtin_amdgcn_global_load_async_to_lds_b128((gv4i_p)g, (lv4i_p)l, 0, 0);
#else
  *(uint4*)l = *(const uint4*)g;
#endif
}

DEV void wait_staged() {
#if HAS_ASYNC_LDS
#if __has_builtin(__builtin_amdgcn_s_wait_asynccnt)
  __builtin_amdgcn_s_wait_asynccnt(0);
#else
  asm volatile("s_wait_asynccnt 0x0" ::: "memory");
#endif
#endif
}

// Round-to-nearest-even f32 -> bf16 (bit pattern as ushort).
DEV unsigned short f2bf(float f) {
  unsigned u = __float_as_uint(f);
  unsigned r = u + 0x7FFFu + ((u >> 16) & 1u);
  return (unsigned short)(r >> 16);
}

union FragBF { bf16x16 v; unsigned u[8]; };

// A-matrix fragment (16x32 bf16, MxK). row32 = &A[lane&15][chunk*32] as u32.
// ISA: lanes 0-15: VGPR0-3 K=0..7, VGPR4-7 K=16..23; lanes 16-31: +8 on K.
DEV bf16x16 load_a_frag(const unsigned* row32, int hi) {
  FragBF f;
  const int h = hi ? 4 : 0;
#pragma unroll
  for (int v = 0; v < 4; ++v) {
    f.u[v]     = row32[v + h];
    f.u[v + 4] = row32[v + 8 + h];
  }
  return f.v;
}

// B-matrix fragment (32x16 bf16, KxN) from column-major storage Bt[n][k].
// Same K order as A: lanes 0-15: VGPR0-3 K=0..7, VGPR4-7 K=16..23; lanes 16-31: +8 on K.
DEV bf16x16 load_b_frag(const unsigned* row32, int hi) {
  FragBF f;
  const int h = hi ? 4 : 0;
#pragma unroll
  for (int v = 0; v < 4; ++v) {
    f.u[v]     = row32[v + h];
    f.u[v + 4] = row32[v + 8 + h];
  }
  return f.v;
}

DEV f32x8 wmma_bf16(bf16x16 a, bf16x16 b, f32x8 c) {
  return __builtin_amdgcn_wmma_f32_16x16x32_bf16(
      false, a, false, b, (short)0, c, false, false);
}

// ---------------------------------------------------------------------------
// Kernel 1: QKV projection.  grid = (ROWS/64, 3), block = 128 (4 waves).
// y==0 -> Q row-major; y==1 -> K row-major; y==2 -> V stored transposed
// per batch as vbt[b][e][s] so the attention kernel can stage V^T tiles as
// pure row copies (async-LDS friendly, no in-kernel transpose).
// ---------------------------------------------------------------------------
__global__ __launch_bounds__(128)
void qkv_proj_kernel(const float* __restrict__ x,
                     const float* __restrict__ w,
                     unsigned short* __restrict__ qkv) {
  __shared__ __attribute__((aligned(16))) unsigned short xs[64 * D_IN];  // 32 KB
  __shared__ __attribute__((aligned(16))) unsigned short wt[64 * D_IN];  // 32 KB (W^T)

  const int tid  = threadIdx.x;
  const int wv   = tid >> 5;
  const int lane = tid & 31;
  const int l15  = lane & 15;
  const int hi   = lane >> 4;
  const int rowBase = blockIdx.x * 64;
  const int y       = blockIdx.y;        // 0=Q, 1=K, 2=V

  // Stage x tile [64][256] f32 -> bf16 LDS (row-major).
  const float4* x4 = (const float4*)(x + (size_t)rowBase * D_IN);
#pragma unroll
  for (int i = 0; i < 32; ++i) {
    int idx = tid + i * 128;             // 0..4095 float4s
    int r   = idx >> 6;
    int c4  = idx & 63;
    float4 t = x4[idx];
    unsigned short* p = &xs[r * D_IN + c4 * 4];
    p[0] = f2bf(t.x); p[1] = f2bf(t.y); p[2] = f2bf(t.z); p[3] = f2bf(t.w);
  }
  // Stage W[y] [256][64] transposed -> wt[n][k] bf16.
  const float4* w4 = (const float4*)(w + (size_t)y * D_IN * D_OUT);
#pragma unroll
  for (int i = 0; i < 32; ++i) {
    int idx = tid + i * 128;
    int k   = idx >> 4;
    int n0  = (idx & 15) * 4;
    float4 t = w4[idx];
    wt[(n0 + 0) * D_IN + k] = f2bf(t.x);
    wt[(n0 + 1) * D_IN + k] = f2bf(t.y);
    wt[(n0 + 2) * D_IN + k] = f2bf(t.z);
    wt[(n0 + 3) * D_IN + k] = f2bf(t.w);
  }
  __syncthreads();

  const f32x8 zero = {0.f, 0.f, 0.f, 0.f, 0.f, 0.f, 0.f, 0.f};
  f32x8 acc[4] = {zero, zero, zero, zero};

  const unsigned* arow = (const unsigned*)&xs[(wv * 16 + l15) * D_IN];
#pragma unroll
  for (int kk = 0; kk < 8; ++kk) {                    // 8 chunks of K=32
    bf16x16 a = load_a_frag(arow + kk * 16, hi);
    bf16x16 bfr[4];
#pragma unroll
    for (int c = 0; c < 4; ++c) {
      const unsigned* brow = (const unsigned*)&wt[(c * 16 + l15) * D_IN];
      bfr[c] = load_b_frag(brow + kk * 16, hi);
    }
#pragma unroll
    for (int c = 0; c < 4; ++c) acc[c] = wmma_bf16(a, bfr[c], acc[c]);
  }

  if (y == 2) {
    // V^T store: vbt[b][e][s], s within this block's batch.
    const int bidx = rowBase >> 12;      // rowBase / 4096
    const int s0   = rowBase & 4095;
    unsigned short* dst = qkv + 2 * (size_t)QKV_ELTS + (size_t)bidx * D_OUT * S_LEN;
#pragma unroll
    for (int c = 0; c < 4; ++c)
#pragma unroll
      for (int r = 0; r < 8; ++r) {
        int m = wv * 16 + r + hi * 8;
        int e = c * 16 + l15;
        dst[(size_t)e * S_LEN + s0 + m] = f2bf(acc[c][r]);
      }
  } else {
    unsigned short* dst = qkv + (size_t)y * QKV_ELTS;
#pragma unroll
    for (int c = 0; c < 4; ++c)
#pragma unroll
      for (int r = 0; r < 8; ++r) {
        int m = wv * 16 + r + hi * 8;
        dst[(size_t)(rowBase + m) * D_OUT + c * 16 + l15] = f2bf(acc[c][r]);
      }
  }
}

// ---------------------------------------------------------------------------
// Kernel 2: flash attention.  grid = (S/64, B), block = 128 (4 waves).
// Double-buffered K (row-major) and V^T tiles staged async into LDS.
// ---------------------------------------------------------------------------
__global__ __launch_bounds__(128)
void flash_attn_kernel(const unsigned short* __restrict__ qb,
                       const unsigned short* __restrict__ kb,
                       const unsigned short* __restrict__ vbt,
                       float* __restrict__ out) {
  __shared__ __attribute__((aligned(16))) unsigned short Ks[2][64 * 64];   // 16 KB
  __shared__ __attribute__((aligned(16))) unsigned short Vt[2][64 * 64];   // 16 KB
  __shared__ __attribute__((aligned(16))) unsigned short Ps[4 * 16 * 64];  //  8 KB

  const int tid  = threadIdx.x;
  const int wv   = tid >> 5;
  const int lane = tid & 31;
  const int l15  = lane & 15;
  const int hi   = lane >> 4;
  const int b    = blockIdx.y;
  const int qrow0 = blockIdx.x * 64;

  const unsigned short* kbase  = kb  + (size_t)b * S_LEN * D_OUT;  // [s][e]
  const unsigned short* vtbase = vbt + (size_t)b * D_OUT * S_LEN;  // [e][s]

  // Q strip A-fragments (head dim 64 = 2 K-chunks), loaded once from global.
  const unsigned* qrow =
      (const unsigned*)(qb + ((size_t)b * S_LEN + qrow0 + wv * 16 + l15) * D_OUT);
  bf16x16 qa0 = load_a_frag(qrow, hi);
  bf16x16 qa1 = load_a_frag(qrow + 16, hi);

  const f32x8 zero = {0.f, 0.f, 0.f, 0.f, 0.f, 0.f, 0.f, 0.f};
  f32x8 o[4] = {zero, zero, zero, zero};
  float mrow[8], lrow[8];
#pragma unroll
  for (int r = 0; r < 8; ++r) { mrow[r] = -__builtin_inff(); lrow[r] = 0.f; }

  // Stage KV tile jj into buffer buf (8 x b128 per thread, all row copies).
  auto stage_tile = [&](int jj, int buf) {
    const unsigned short* kg = kbase + (size_t)jj * 64 * D_OUT;   // 8 KB contiguous
#pragma unroll
    for (int i = 0; i < 4; ++i) {
      int c = tid + i * 128;                       // 512 16B chunks
      copy16_to_lds(kg + c * 8, &Ks[buf][c * 8]);
    }
    const unsigned short* vg = vtbase + jj * 64;   // 64 rows of 128 B, stride S_LEN
#pragma unroll
    for (int i = 0; i < 4; ++i) {
      int c = tid + i * 128;
      int e = c >> 3;
      int o16 = (c & 7) * 8;
      copy16_to_lds(vg + (size_t)e * S_LEN + o16, &Vt[buf][e * 64 + o16]);
    }
  };

  stage_tile(0, 0);

  for (int j = 0; j < S_LEN / 64; ++j) {
    const int cur = j & 1;
    wait_staged();          // own async tile complete (ASYNCcnt==0)
    __syncthreads();        // all waves' tiles complete; prev compute done
    if (j + 1 < S_LEN / 64) stage_tile(j + 1, cur ^ 1);  // prefetch next tile

    // ---- S = Q . K^T : batch all 8 B-fragments, then 8 back-to-back WMMAs.
    bf16x16 kb0[4], kb1[4];
#pragma unroll
    for (int c = 0; c < 4; ++c) {
      const unsigned* krow = (const unsigned*)&Ks[cur][(c * 16 + l15) * D_OUT];
      kb0[c] = load_b_frag(krow, hi);
      kb1[c] = load_b_frag(krow + 16, hi);
    }
    f32x8 s[4];
#pragma unroll
    for (int c = 0; c < 4; ++c) s[c] = wmma_bf16(qa0, kb0[c], zero);
#pragma unroll
    for (int c = 0; c < 4; ++c) s[c] = wmma_bf16(qa1, kb1[c], s[c]);

    // ---- Online softmax (row = r + hi*8; row elements across 16 lanes).
#pragma unroll
    for (int r = 0; r < 8; ++r) {
      float sv[4];
      float lm = -__builtin_inff();
#pragma unroll
      for (int c = 0; c < 4; ++c) { sv[c] = s[c][r] * SCALE; lm = fmaxf(lm, sv[c]); }
#pragma unroll
      for (int off = 1; off <= 8; off <<= 1) lm = fmaxf(lm, __shfl_xor(lm, off, 32));
      float mnew  = fmaxf(mrow[r], lm);
      float alpha = __expf(mrow[r] - mnew);
      float ls = 0.f;
      unsigned short* prow = &Ps[wv * 1024 + (r + hi * 8) * 64];
#pragma unroll
      for (int c = 0; c < 4; ++c) {
        float p = __expf(sv[c] - mnew);
        ls += p;
        prow[c * 16 + l15] = f2bf(p);
      }
#pragma unroll
      for (int off = 1; off <= 8; off <<= 1) ls += __shfl_xor(ls, off, 32);
      lrow[r] = lrow[r] * alpha + ls;
      mrow[r] = mnew;
#pragma unroll
      for (int c = 0; c < 4; ++c) o[c][r] *= alpha;
    }

    // ---- O += P . V  (P strip wave-private; same-wave DS ordering).
    const unsigned* prow32 = (const unsigned*)&Ps[wv * 1024 + l15 * 64];
    bf16x16 pa0 = load_a_frag(prow32, hi);
    bf16x16 pa1 = load_a_frag(prow32 + 16, hi);
    bf16x16 vbf[4];
#pragma unroll
    for (int c = 0; c < 4; ++c)
      vbf[c] = load_b_frag((const unsigned*)&Vt[cur][(c * 16 + l15) * 64], hi);
#pragma unroll
    for (int c = 0; c < 4; ++c) o[c] = wmma_bf16(pa0, vbf[c], o[c]);
#pragma unroll
    for (int c = 0; c < 4; ++c)
      vbf[c] = load_b_frag((const unsigned*)&Vt[cur][(c * 16 + l15) * 64] + 16, hi);
#pragma unroll
    for (int c = 0; c < 4; ++c) o[c] = wmma_bf16(pa1, vbf[c], o[c]);
  }

  // ---- Normalize and write fp32 output.
  float* op = out + ((size_t)b * S_LEN + qrow0) * D_OUT;
#pragma unroll
  for (int r = 0; r < 8; ++r) {
    float inv = 1.f / lrow[r];
#pragma unroll
    for (int c = 0; c < 4; ++c)
      op[(size_t)(wv * 16 + r + hi * 8) * D_OUT + c * 16 + l15] = o[c][r] * inv;
  }
}

// ---------------------------------------------------------------------------
extern "C" void kernel_launch(void* const* d_in, const int* in_sizes, int n_in,
                              void* d_out, int out_size, void* d_ws, size_t ws_size,
                              hipStream_t stream) {
  const float* x = (const float*)d_in[0];   // [4,4096,256]
  const float* w = (const float*)d_in[1];   // [3,256,64]
  float* out = (float*)d_out;               // [4,4096,64]

  unsigned short* qkv = (unsigned short*)d_ws;   // Q | K | V^T, bf16, 2 MB each
  unsigned short* qb  = qkv;
  unsigned short* kb  = qkv + QKV_ELTS;
  unsigned short* vbt = qkv + 2 * (size_t)QKV_ELTS;

  dim3 g1(ROWS / 64, 3);
  qkv_proj_kernel<<<g1, 128, 0, stream>>>(x, w, qkv);

  dim3 g2(S_LEN / 64, BATCH);
  flash_attn_kernel<<<g2, 128, 0, stream>>>(qb, kb, vbt, out);
}
